// Autoformer_90099823935832
// MI455X (gfx1250) — matrix operand order repaired and checked by reading, not compiled, not run on hardware
//
#include <hip/hip_runtime.h>
#include <math.h>

// ---------------- problem constants (from reference) ----------------
constexpr int BATCH  = 32;
constexpr int SEQL   = 720;
constexpr int LABL   = 336;
constexpr int PREDL  = 720;
constexpr int DECLEN = LABL + PREDL;   // 1056
constexpr int CINV   = 21;             // enc_in / dec_in / c_out
constexpr int DM     = 512;            // d_model
constexpr int DFFN   = 2048;
constexpr int TFD    = 4;              // time features
constexpr int ELN    = 2;              // encoder layers
constexpr int TOPK   = 6;              // int(1*log(720)) == int(1*log(1056)) == 6

// ---------------- CDNA5 WMMA types ----------------
typedef __bf16 bf16x16 __attribute__((ext_vector_type(16)));
typedef float  f32x8   __attribute__((ext_vector_type(8)));

__device__ __forceinline__ unsigned pack2(float lo, float hi) {
  // let the compiler pick the native f32->bf16 convert
  union { __bf16 h[2]; unsigned u; } o;
  o.h[0] = (__bf16)lo; o.h[1] = (__bf16)hi;
  return o.u;
}
__device__ __forceinline__ uint4 pack8(float4 a, float4 b) {
  uint4 r;
  r.x = pack2(a.x, a.y); r.y = pack2(a.z, a.w);
  r.z = pack2(b.x, b.y); r.w = pack2(b.z, b.w);
  return r;
}

union FragU { uint4 u[2]; bf16x16 v; };

__device__ __forceinline__ float gelu_exact(float x) {
  return 0.5f * x * (1.0f + erff(x * 0.70710678118654752f));
}

// ---------------------------------------------------------------
// GEMM: C[M,N] = act(A[M,K] @ W[N,K]^T + bias[N] (+ resid[M,N]))
// bf16 WMMA, fp32 accumulate. Block = 256 thr = 8 wave32, 64x64 tile,
// each wave computes two 16x16 WMMA tiles sharing one A fragment.
// LDS tiles are stored in *fragment layout* (ISA 7.12.2) so each
// fragment is reassembled with two ds_load_b128 per matrix:
//   A frag: lane = (row%16) + 16*half ; 8-elem chunk cblk in 0..3 ->
//           lane-half = cblk&1, uint4-chunk = cblk>>1
//   B frag: lane = (col%16) + 16*((K>>3)&1) ; lane holds two 8-K runs
//           of one W row -> kchunk kc in 0..3 -> half = kc&1, chunk = kc>>1
// LDS is double-buffered (one barrier per K-step); next tile's global
// loads are issued before the WMMAs so their latency hides behind the
// matrix ops. M % 64 == 0 and K % 32 == 0 guaranteed by caller; N guarded.
// ---------------------------------------------------------------
__global__ __launch_bounds__(256)
void gemm_wmma_kernel(const float* __restrict__ A, const float* __restrict__ W,
                      const float* __restrict__ bias, const float* __restrict__ resid,
                      float* __restrict__ C, int M, int N, int K, int act) {
  __shared__ uint4 AsF[2][4][32][2];   // [buf][row-tile][lane][uint4-chunk]
  __shared__ uint4 BsF[2][4][32][2];   // [buf][col-tile][lane][uint4-chunk]
  const int tid  = threadIdx.x;
  const int lane = tid & 31;
  const int wid  = tid >> 5;
  const int m0 = blockIdx.x * 64;
  const int n0 = blockIdx.y * 64;
  const int tr  = wid & 3;          // wave's 16-row tile within 64
  const int tc0 = (wid >> 2) * 2;   // wave's first 16-col tile (0 or 2)

  f32x8 acc0 = {0.f,0.f,0.f,0.f,0.f,0.f,0.f,0.f};
  f32x8 acc1 = {0.f,0.f,0.f,0.f,0.f,0.f,0.f,0.f};

  // --- cooperative loader assignments (one A chunk + one B chunk per thread)
  const int ar  = tid >> 2, acb = tid & 3;                 // A: row 0..63, chunk 0..3
  const float* aptr = A + (size_t)(m0 + ar) * K + acb * 8;
  const int a_st = ar >> 4;                                // dest row-tile
  const int a_ln = (ar & 15) + ((acb & 1) << 4);           // dest lane
  const int a_ch = acb >> 1;                               // dest uint4 chunk

  const int bn  = tid >> 2, bkc = tid & 3;                 // B: col 0..63, kchunk 0..3
  const int gn  = n0 + bn;
  const float* bptr = W + (size_t)gn * K + bkc * 8;
  const int b_ct = bn >> 4;                                // dest col-tile
  const int b_ln = (bn & 15) + ((bkc & 1) << 4);           // dest lane
  const int b_ch = bkc >> 1;                               // dest uint4 chunk
  const bool bok = (gn < N);

  // --- preload K-step 0 into buffer 0
  {
    const float4* ap = (const float4*)aptr;
    AsF[0][a_st][a_ln][a_ch] = pack8(ap[0], ap[1]);
    float4 z = {0.f,0.f,0.f,0.f}, b0 = z, b1 = z;
    if (bok) { const float4* bp = (const float4*)bptr; b0 = bp[0]; b1 = bp[1]; }
    BsF[0][b_ct][b_ln][b_ch] = pack8(b0, b1);
  }
  __syncthreads();

  for (int k0 = 0; k0 < K; k0 += 32) {
    const int cur = (k0 >> 5) & 1, nxt = cur ^ 1;
    const bool hasnext = (k0 + 32) < K;

    // 1) fragment loads from current buffer (ds_load_b128 x6)
    FragU af, bf0, bf1;
    af.u[0]  = AsF[cur][tr][lane][0];      af.u[1]  = AsF[cur][tr][lane][1];
    bf0.u[0] = BsF[cur][tc0][lane][0];     bf0.u[1] = BsF[cur][tc0][lane][1];
    bf1.u[0] = BsF[cur][tc0 + 1][lane][0]; bf1.u[1] = BsF[cur][tc0 + 1][lane][1];

    // 2) issue next tile's global loads (latency hidden behind WMMAs)
    float4 av0, av1, bv0 = {0.f,0.f,0.f,0.f}, bv1 = {0.f,0.f,0.f,0.f};
    if (hasnext) {
      const float4* ap = (const float4*)(aptr + k0 + 32);
      av0 = ap[0]; av1 = ap[1];
      if (bok) { const float4* bp = (const float4*)(bptr + k0 + 32); bv0 = bp[0]; bv1 = bp[1]; }
      if (k0 + 64 < K) __builtin_prefetch(aptr + k0 + 64, 0, 3);   // global_prefetch_b8
    }

    // 3) matrix ops
    acc0 = __builtin_amdgcn_wmma_f32_16x16x32_bf16(
        false, af.v, false, bf0.v, (short)0, acc0, false, false);
    acc1 = __builtin_amdgcn_wmma_f32_16x16x32_bf16(
        false, af.v, false, bf1.v, (short)0, acc1, false, false);

    // 4) pack and stage next tile, single barrier per step
    if (hasnext) {
      AsF[nxt][a_st][a_ln][a_ch] = pack8(av0, av1);
      BsF[nxt][b_ct][b_ln][b_ch] = pack8(bv0, bv1);
    }
    __syncthreads();
  }

  // C/D layout: VGPR i -> row = i + 8*(lane>=16), col = lane%16
  const int rbase = m0 + tr * 16 + ((lane >= 16) ? 8 : 0);
  #pragma unroll
  for (int half = 0; half < 2; ++half) {
    const f32x8& acc = half ? acc1 : acc0;
    const int col = n0 + (tc0 + half) * 16 + (lane & 15);
    if (col < N) {
      const float bv = bias ? bias[col] : 0.f;
      #pragma unroll
      for (int i = 0; i < 8; ++i) {
        int row = rbase + i;
        float v = acc[i] + bv;
        if (resid) v += resid[(size_t)row * N + col];
        if (act == 1) v = gelu_exact(v);
        C[(size_t)row * N + col] = v;
      }
    }
  }
}

// ---------------------------------------------------------------
// Circular conv3 embedding: out[b,l,o] = sum_{c,j} x[b,(l+j-1)%L,c]*W[o,c,j]
//   (+ mark[b,l,:] @ timeW[o,:])  (+ resid[b,l,o])
// ---------------------------------------------------------------
__global__ void conv3_kernel(const float* __restrict__ x, const float* __restrict__ W,
                             const float* __restrict__ timeW, const float* __restrict__ mark,
                             const float* __restrict__ resid, float* __restrict__ out,
                             int B_, int L, int Cin, int Cout) {
  size_t i = blockIdx.x * (size_t)blockDim.x + threadIdx.x;
  size_t n = (size_t)B_ * L * Cout;
  if (i >= n) return;
  int o = (int)(i % Cout);
  size_t bl = i / Cout;
  int l = (int)(bl % L);
  int b = (int)(bl / L);
  const float* xb = x + (size_t)b * L * Cin;
  int lm = (l - 1 + L) % L, lp = (l + 1) % L;
  const float* x0 = xb + (size_t)lm * Cin;
  const float* x1 = xb + (size_t)l  * Cin;
  const float* x2 = xb + (size_t)lp * Cin;
  const float* w  = W + (size_t)o * Cin * 3;
  float acc = 0.f;
  for (int c = 0; c < Cin; ++c)
    acc += x0[c] * w[c*3+0] + x1[c] * w[c*3+1] + x2[c] * w[c*3+2];
  if (timeW) {
    const float* mb = mark + ((size_t)b * L + l) * TFD;
    const float* tw = timeW + (size_t)o * TFD;
    #pragma unroll
    for (int t = 0; t < TFD; ++t) acc += mb[t] * tw[t];
  }
  if (resid) acc += resid[i];
  out[i] = acc;
}

// ---------------------------------------------------------------
// Series decomposition: moving avg k=25, replicate padding.
// seas = x - m ; trend = m (store or accumulate). Either out may be null.
// ---------------------------------------------------------------
__global__ void decomp_kernel(const float* __restrict__ x, float* __restrict__ seas,
                              float* __restrict__ trend, int accum,
                              int B_, int L, int C) {
  size_t i = blockIdx.x * (size_t)blockDim.x + threadIdx.x;
  size_t n = (size_t)B_ * L * C;
  if (i >= n) return;
  int c = (int)(i % C);
  size_t bl = i / C;
  int l = (int)(bl % L);
  int b = (int)(bl / L);
  const float* xb = x + (size_t)b * L * C + c;
  float s = 0.f;
  #pragma unroll
  for (int d = -12; d <= 12; ++d) {
    int li = l + d;
    li = li < 0 ? 0 : (li >= L ? L - 1 : li);
    s += xb[(size_t)li * C];
  }
  float m = s * (1.0f / 25.0f);
  if (seas)  seas[i] = x[i] - m;
  if (trend) { if (accum) trend[i] += m; else trend[i] = m; }
}

// ---------------------------------------------------------------
// LayerNorm over D=512, one wave32 per row (8 rows per 256-thr block)
// ---------------------------------------------------------------
__global__ __launch_bounds__(256)
void layernorm_kernel(const float* __restrict__ x, const float* __restrict__ w,
                      const float* __restrict__ bb, float* __restrict__ out, int rows) {
  int row  = blockIdx.x * 8 + (threadIdx.x >> 5);
  int lane = threadIdx.x & 31;
  if (row >= rows) return;
  const float* xr = x + (size_t)row * DM + lane * 16;
  float vbuf[16];
  float s = 0.f, s2 = 0.f;
  #pragma unroll
  for (int j = 0; j < 16; ++j) { float t = xr[j]; vbuf[j] = t; s += t; s2 += t * t; }
  #pragma unroll
  for (int off = 16; off > 0; off >>= 1) {
    s  += __shfl_xor(s,  off, 32);
    s2 += __shfl_xor(s2, off, 32);
  }
  float mu  = s  * (1.f / DM);
  float var = s2 * (1.f / DM) - mu * mu;
  float r   = rsqrtf(var + 1e-5f);
  float* orow = out + (size_t)row * DM + lane * 16;
  const float* wr = w + lane * 16;
  const float* br = bb + lane * 16;
  #pragma unroll
  for (int j = 0; j < 16; ++j) orow[j] = (vbuf[j] - mu) * r * wr[j] + br[j];
}

// ---------------------------------------------------------------
// Time-domain circular correlation (== irfft(rfft(q)*conj(rfft(k))),
// channel-averaged): corr[b,tau] = (1/D) sum_t <q[b,t,:], kpad[b,(t-tau)%L,:]>
// kpad rows >= S are zero. Block per (tau, b), 256-thread reduction.
// Per-batch Q/K tiles (~3MB) stay resident in the 192MB L2 across taus.
// ---------------------------------------------------------------
__global__ __launch_bounds__(256)
void corr_kernel(const float* __restrict__ q, const float* __restrict__ k,
                 float* __restrict__ corr, int L, int S) {
  int tau = blockIdx.x, b = blockIdx.y;
  const float4* qb = (const float4*)(q + (size_t)b * L * DM);
  const float4* kb = (const float4*)(k + (size_t)b * S * DM);
  float p = 0.f;
  for (int t = threadIdx.x; t < L; t += 256) {
    int ts = t - tau; if (ts < 0) ts += L;
    if (ts < S) {
      const float4* qr = qb + (size_t)t  * (DM / 4);
      const float4* kr = kb + (size_t)ts * (DM / 4);
      #pragma unroll 4
      for (int c = 0; c < DM / 4; ++c) {
        float4 a = qr[c], d = kr[c];
        p += a.x * d.x + a.y * d.y + a.z * d.z + a.w * d.w;
      }
    }
  }
  __shared__ float red[256];
  red[threadIdx.x] = p; __syncthreads();
  for (int s = 128; s > 0; s >>= 1) {
    if (threadIdx.x < s) red[threadIdx.x] += red[threadIdx.x + s];
    __syncthreads();
  }
  if (threadIdx.x == 0) corr[(size_t)b * L + tau] = red[0] * (1.0f / DM);
}

// ---------------------------------------------------------------
// Top-6 lags + softmax weights, one wave32 per batch.
// ---------------------------------------------------------------
__global__ void topk_kernel(const float* __restrict__ corr, float* __restrict__ tw,
                            int* __restrict__ delay, int L) {
  int b = blockIdx.x, lane = threadIdx.x;
  const float* cb = corr + (size_t)b * L;
  __shared__ int   sidx[TOPK];
  __shared__ float sval[TOPK];
  for (int kk = 0; kk < TOPK; ++kk) {
    float bv = -1e30f; int bi = 0;
    for (int t = lane; t < L; t += 32) {
      bool used = false;
      for (int j = 0; j < kk; ++j) used |= (sidx[j] == t);
      float v = cb[t];
      if (!used && v > bv) { bv = v; bi = t; }
    }
    #pragma unroll
    for (int off = 16; off > 0; off >>= 1) {
      float ov = __shfl_down(bv, off, 32);
      int   oi = __shfl_down(bi, off, 32);
      if (ov > bv) { bv = ov; bi = oi; }
    }
    if (lane == 0) { sidx[kk] = bi; sval[kk] = bv; }
    __syncthreads();
  }
  if (lane == 0) {
    float mx = sval[0];
    for (int j = 1; j < TOPK; ++j) mx = fmaxf(mx, sval[j]);
    float e[TOPK]; float se = 0.f;
    for (int j = 0; j < TOPK; ++j) { e[j] = __expf(sval[j] - mx); se += e[j]; }
    for (int j = 0; j < TOPK; ++j) {
      tw[b * TOPK + j] = e[j] / se;
      delay[b * TOPK + j] = sidx[j];
    }
  }
}

// ---------------------------------------------------------------
// Delay aggregation: out[b,l,:] = sum_k w[b,k] * vpad[b,(l+delay_k)%L,:]
// ---------------------------------------------------------------
__global__ void aggregate_kernel(const float* __restrict__ v, const float* __restrict__ tw,
                                 const int* __restrict__ delay, float* __restrict__ out,
                                 int L, int S, size_t n) {
  size_t i = blockIdx.x * (size_t)blockDim.x + threadIdx.x;
  if (i >= n) return;
  int c4 = (int)(i % (DM / 4));
  size_t bl = i / (DM / 4);
  int l = (int)(bl % L);
  int b = (int)(bl / L);
  float4 acc = {0.f, 0.f, 0.f, 0.f};
  #pragma unroll
  for (int kk = 0; kk < TOPK; ++kk) {
    int idx = l + delay[b * TOPK + kk];
    if (idx >= L) idx -= L;
    if (idx < S) {
      float w = tw[b * TOPK + kk];
      float4 val = ((const float4*)(v + ((size_t)b * S + idx) * DM))[c4];
      acc.x += w * val.x; acc.y += w * val.y;
      acc.z += w * val.z; acc.w += w * val.w;
    }
  }
  ((float4*)(out + ((size_t)b * L + l) * DM))[c4] = acc;
}

// ---------------------------------------------------------------
// misc elementwise
// ---------------------------------------------------------------
__global__ void add_kernel(const float* __restrict__ a, const float* __restrict__ b,
                           float* __restrict__ o, size_t n) {
  size_t i = blockIdx.x * (size_t)blockDim.x + threadIdx.x;
  if (i < n) o[i] = a[i] + b[i];
}

__global__ void mean_time_kernel(const float* __restrict__ x, float* __restrict__ out,
                                 int B_, int L, int C) {
  int i = blockIdx.x * blockDim.x + threadIdx.x;
  if (i >= B_ * C) return;
  int c = i % C, b = i / C;
  const float* xb = x + (size_t)b * L * C + c;
  float s = 0.f;
  for (int l = 0; l < L; ++l) s += xb[(size_t)l * C];
  out[i] = s / L;
}

__global__ void build_init_kernel(const float* __restrict__ se, const float* __restrict__ te,
                                  const float* __restrict__ mean, float* __restrict__ si,
                                  float* __restrict__ ti) {
  size_t i = blockIdx.x * (size_t)blockDim.x + threadIdx.x;
  size_t n = (size_t)BATCH * DECLEN * CINV;
  if (i >= n) return;
  int c = (int)(i % CINV);
  size_t bl = i / CINV;
  int l = (int)(bl % DECLEN);
  int b = (int)(bl / DECLEN);
  if (l < LABL) {
    size_t src = ((size_t)b * SEQL + (SEQL - LABL + l)) * CINV + c;
    si[i] = se[src];
    ti[i] = te[src];
  } else {
    si[i] = 0.f;
    ti[i] = mean[b * CINV + c];
  }
}

__global__ void final_kernel(const float* __restrict__ trend, const float* __restrict__ season,
                             float* __restrict__ out) {
  size_t i = blockIdx.x * (size_t)blockDim.x + threadIdx.x;
  size_t n = (size_t)BATCH * PREDL * CINV;
  if (i >= n) return;
  int c = (int)(i % CINV);
  size_t bp = i / CINV;
  int p = (int)(bp % PREDL);
  int b = (int)(bp / PREDL);
  size_t src = ((size_t)b * DECLEN + (LABL + p)) * CINV + c;
  out[i] = trend[src] + season[src];
}

// ---------------- host-side launch helpers ----------------
static inline void gemm(const float* A, const float* W, const float* bias,
                        const float* resid, float* C, int M, int N, int K, int act,
                        hipStream_t s) {
  dim3 g(M / 64, (N + 63) / 64);
  gemm_wmma_kernel<<<g, 256, 0, s>>>(A, W, bias, resid, C, M, N, K, act);
}

static inline void ew_add(const float* a, const float* b, float* o, size_t n, hipStream_t s) {
  add_kernel<<<(unsigned)((n + 255) / 256), 256, 0, s>>>(a, b, o, n);
}

static void run_attn(const float* xq, int Lq, const float* xkv, int Skv,
                     const float* Wq, const float* bq, const float* Wk, const float* bk,
                     const float* Wv, const float* bv, const float* Wo, const float* bo,
                     float* Qb, float* Kb, float* Vb, float* Ab, float* attn_out,
                     float* corrb, float* twb, int* dlb, hipStream_t s) {
  int Mq = BATCH * Lq, Mkv = BATCH * Skv;
  gemm(xq,  Wq, bq, nullptr, Qb, Mq,  DM, DM, 0, s);
  gemm(xkv, Wk, bk, nullptr, Kb, Mkv, DM, DM, 0, s);
  gemm(xkv, Wv, bv, nullptr, Vb, Mkv, DM, DM, 0, s);
  corr_kernel<<<dim3(Lq, BATCH), 256, 0, s>>>(Qb, Kb, corrb, Lq, Skv);
  topk_kernel<<<BATCH, 32, 0, s>>>(corrb, twb, dlb, Lq);
  size_t na = (size_t)BATCH * Lq * (DM / 4);
  aggregate_kernel<<<(unsigned)((na + 255) / 256), 256, 0, s>>>(Vb, twb, dlb, Ab, Lq, Skv, na);
  gemm(Ab, Wo, bo, nullptr, attn_out, Mq, DM, DM, 0, s);
}

extern "C" void kernel_launch(void* const* d_in, const int* in_sizes, int n_in,
                              void* d_out, int out_size, void* d_ws, size_t ws_size,
                              hipStream_t stream) {
  const float* x_enc      = (const float*)d_in[0];
  const float* x_mark_enc = (const float*)d_in[1];
  const float* x_mark_dec = (const float*)d_in[3];
  const float* enc_tok_W  = (const float*)d_in[4];
  const float* enc_time_W = (const float*)d_in[5];
  const float* dec_tok_W  = (const float*)d_in[6];
  const float* dec_time_W = (const float*)d_in[7];
  const float* enc_Wq = (const float*)d_in[8];
  const float* enc_bq = (const float*)d_in[9];
  const float* enc_Wk = (const float*)d_in[10];
  const float* enc_bk = (const float*)d_in[11];
  const float* enc_Wv = (const float*)d_in[12];
  const float* enc_bv = (const float*)d_in[13];
  const float* enc_Wo = (const float*)d_in[14];
  const float* enc_bo = (const float*)d_in[15];
  const float* enc_W1 = (const float*)d_in[16];
  const float* enc_b1 = (const float*)d_in[17];
  const float* enc_W2 = (const float*)d_in[18];
  const float* enc_b2 = (const float*)d_in[19];
  const float* enc_norm_w = (const float*)d_in[20];
  const float* enc_norm_b = (const float*)d_in[21];
  const float* ds_Wq = (const float*)d_in[22];
  const float* ds_bq = (const float*)d_in[23];
  const float* ds_Wk = (const float*)d_in[24];
  const float* ds_bk = (const float*)d_in[25];
  const float* ds_Wv = (const float*)d_in[26];
  const float* ds_bv = (const float*)d_in[27];
  const float* ds_Wo = (const float*)d_in[28];
  const float* ds_bo = (const float*)d_in[29];
  const float* dc_Wq = (const float*)d_in[30];
  const float* dc_bq = (const float*)d_in[31];
  const float* dc_Wk = (const float*)d_in[32];
  const float* dc_bk = (const float*)d_in[33];
  const float* dc_Wv = (const float*)d_in[34];
  const float* dc_bv = (const float*)d_in[35];
  const float* dc_Wo = (const float*)d_in[36];
  const float* dc_bo = (const float*)d_in[37];
  const float* dec_W1 = (const float*)d_in[38];
  const float* dec_b1 = (const float*)d_in[39];
  const float* dec_W2 = (const float*)d_in[40];
  const float* dec_b2 = (const float*)d_in[41];
  const float* dec_trend_W = (const float*)d_in[42];
  const float* dec_norm_w  = (const float*)d_in[43];
  const float* dec_norm_b  = (const float*)d_in[44];
  const float* dec_proj_W  = (const float*)d_in[45];
  const float* dec_proj_b  = (const float*)d_in[46];
  float* out = (float*)d_out;

  // -------- workspace carving --------
  float* ws = (float*)d_ws;
  size_t off = 0;
  auto alloc = [&](size_t n) { float* p = ws + off; off += n; return p; };
  const size_t s_enc = (size_t)BATCH * SEQL   * DM;
  const size_t s_dec = (size_t)BATCH * DECLEN * DM;
  const size_t s_h   = (size_t)BATCH * DECLEN * DFFN;
  float* encb  = alloc(s_enc);                          // encoder output
  float* Xb    = alloc(s_dec);                          // residual x
  float* Qb    = alloc(s_dec);
  float* Kb    = alloc(s_dec);
  float* Vb    = alloc(s_dec);
  float* Ab    = alloc(s_dec);                          // attn aggregate
  float* Tb    = alloc(s_dec);                          // trend accumulator
  float* Hb    = alloc(s_h);                            // FFN hidden / attn out
  float* corrb = alloc((size_t)BATCH * DECLEN);
  float* twb   = alloc(BATCH * TOPK);
  int*   dlb   = (int*)alloc(BATCH * TOPK);
  float* meanb = alloc(BATCH * CINV);
  float* se_b  = alloc((size_t)BATCH * SEQL * CINV);    // seasonal of x_enc
  float* te_b  = alloc((size_t)BATCH * SEQL * CINV);    // trend of x_enc
  float* ti_b  = alloc((size_t)BATCH * DECLEN * CINV);  // trend_init
  float* si_b  = alloc((size_t)BATCH * DECLEN * CINV);  // seasonal_init
  float* trc_b = alloc((size_t)BATCH * DECLEN * CINV);  // trend after conv
  float* sea_b = alloc((size_t)BATCH * DECLEN * CINV);  // seasonal projection

  const size_t nEnc = s_enc, nDec = s_dec;

  // -------- encoder embedding --------
  {
    size_t n = (size_t)BATCH * SEQL * DM;
    conv3_kernel<<<(unsigned)((n + 255) / 256), 256, 0, stream>>>(
        x_enc, enc_tok_W, enc_time_W, x_mark_enc, nullptr, encb, BATCH, SEQL, CINV, DM);
  }

  // -------- encoder layers --------
  for (int i = 0; i < ELN; ++i) {
    const size_t wsz = (size_t)DM * DM, fsz = (size_t)DFFN * DM;
    run_attn(encb, SEQL, encb, SEQL,
             enc_Wq + i * wsz, enc_bq + i * DM, enc_Wk + i * wsz, enc_bk + i * DM,
             enc_Wv + i * wsz, enc_bv + i * DM, enc_Wo + i * wsz, enc_bo + i * DM,
             Qb, Kb, Vb, Ab, /*attn_out=*/Xb, corrb, twb, dlb, stream);
    ew_add(encb, Xb, Kb, nEnc, stream);
    decomp_kernel<<<(unsigned)((nEnc + 255) / 256), 256, 0, stream>>>(
        Kb, Xb, nullptr, 0, BATCH, SEQL, DM);
    gemm(Xb, enc_W1 + i * fsz, enc_b1 + i * DFFN, nullptr, Hb,
         BATCH * SEQL, DFFN, DM, /*gelu*/1, stream);
    gemm(Hb, enc_W2 + i * fsz, enc_b2 + i * DM, nullptr, Qb,
         BATCH * SEQL, DM, DFFN, 0, stream);
    ew_add(Xb, Qb, Kb, nEnc, stream);
    decomp_kernel<<<(unsigned)((nEnc + 255) / 256), 256, 0, stream>>>(
        Kb, encb, nullptr, 0, BATCH, SEQL, DM);
  }
  layernorm_kernel<<<(BATCH * SEQL) / 8, 256, 0, stream>>>(
      encb, enc_norm_w, enc_norm_b, encb, BATCH * SEQL);

  // -------- decoder init (trend/seasonal of raw x_enc) --------
  mean_time_kernel<<<(BATCH * CINV + 255) / 256, 256, 0, stream>>>(
      x_enc, meanb, BATCH, SEQL, CINV);
  {
    size_t n = (size_t)BATCH * SEQL * CINV;
    decomp_kernel<<<(unsigned)((n + 255) / 256), 256, 0, stream>>>(
        x_enc, se_b, te_b, 0, BATCH, SEQL, CINV);
  }
  {
    size_t n = (size_t)BATCH * DECLEN * CINV;
    build_init_kernel<<<(unsigned)((n + 255) / 256), 256, 0, stream>>>(
        se_b, te_b, meanb, si_b, ti_b);
  }
  {
    size_t n = (size_t)BATCH * DECLEN * DM;
    conv3_kernel<<<(unsigned)((n + 255) / 256), 256, 0, stream>>>(
        si_b, dec_tok_W, dec_time_W, x_mark_dec, nullptr, Xb, BATCH, DECLEN, CINV, DM);
  }

  // -------- decoder: self-attention --------
  run_attn(Xb, DECLEN, Xb, DECLEN,
           ds_Wq, ds_bq, ds_Wk, ds_bk, ds_Wv, ds_bv, ds_Wo, ds_bo,
           Qb, Kb, Vb, Ab, /*attn_out=*/Hb, corrb, twb, dlb, stream);
  ew_add(Xb, Hb, Kb, nDec, stream);
  decomp_kernel<<<(unsigned)((nDec + 255) / 256), 256, 0, stream>>>(
      Kb, Xb, Tb, /*accum=*/0, BATCH, DECLEN, DM);

  // -------- decoder: cross-attention (k/v from encoder) --------
  run_attn(Xb, DECLEN, encb, SEQL,
           dc_Wq, dc_bq, dc_Wk, dc_bk, dc_Wv, dc_bv, dc_Wo, dc_bo,
           Qb, Kb, Vb, Ab, /*attn_out=*/Hb, corrb, twb, dlb, stream);
  ew_add(Xb, Hb, Kb, nDec, stream);
  decomp_kernel<<<(unsigned)((nDec + 255) / 256), 256, 0, stream>>>(
      Kb, Xb, Tb, /*accum=*/1, BATCH, DECLEN, DM);

  // -------- decoder: FFN --------
  gemm(Xb, dec_W1, dec_b1, nullptr, Hb, BATCH * DECLEN, DFFN, DM, /*gelu*/1, stream);
  gemm(Hb, dec_W2, dec_b2, nullptr, Qb, BATCH * DECLEN, DM, DFFN, 0, stream);
  ew_add(Xb, Qb, Kb, nDec, stream);
  decomp_kernel<<<(unsigned)((nDec + 255) / 256), 256, 0, stream>>>(
      Kb, Xb, Tb, /*accum=*/1, BATCH, DECLEN, DM);

  // -------- trend branch: conv3(t1+t2+t3) + trend_init --------
  {
    size_t n = (size_t)BATCH * DECLEN * CINV;
    conv3_kernel<<<(unsigned)((n + 255) / 256), 256, 0, stream>>>(
        Tb, dec_trend_W, nullptr, nullptr, ti_b, trc_b, BATCH, DECLEN, DM, CINV);
  }

  // -------- seasonal branch: layernorm + projection --------
  layernorm_kernel<<<(BATCH * DECLEN) / 8, 256, 0, stream>>>(
      Xb, dec_norm_w, dec_norm_b, Xb, BATCH * DECLEN);
  gemm(Xb, dec_proj_W, dec_proj_b, nullptr, sea_b, BATCH * DECLEN, CINV, DM, 0, stream);

  // -------- final: (trend + seasonal)[:, -PRED:] --------
  {
    size_t n = (size_t)BATCH * PREDL * CINV;
    final_kernel<<<(unsigned)((n + 255) / 256), 256, 0, stream>>>(trc_b, sea_b, out);
  }
}
